// LearnedSimulator_54966991454741
// MI455X (gfx1250) — matrix operand order repaired and checked by reading, not compiled, not run on hardware
//
#include <hip/hip_runtime.h>

// ---------------------------------------------------------------------------
// GNS forward for MI455X (gfx1250): bf16 WMMA GEMMs (16x16x32), f32 accum,
// f32 master state + bf16 operand mirrors. Edge MLP dominates (~525 GFLOP of
// ~575 total); all GEMMs routed through one templated WMMA kernel with fused
// gathers, async-to-LDS weight staging, 32x128 wave tiles and register-
// resident B fragments so WMMAs issue back-to-back after one dscnt wait.
// ---------------------------------------------------------------------------

#define LAT   128
#define RCONN 0.05f

typedef unsigned int   u32;
typedef unsigned short u16;
typedef __attribute__((ext_vector_type(16))) __bf16 v16bf;
typedef __attribute__((ext_vector_type(8)))  float  v8f;
typedef __attribute__((ext_vector_type(4)))  u32    v4u;

#define GM_PLAIN 0   // A rows from A0[m*Ktot]
#define GM_EDGE  1   // K=384: [x[recv[m]], x[snd[m]], e[m]] each 128 wide
#define GM_NODE  2   // K=256: [x[m], agg[m]]

#define LN_SET  0    // state  = ln(in)
#define LN_EDGE 1    // enew   = ln(in); e += enew  (f32 master + bf16 mirror)
#define LN_NODE 2    // x     += ln(in)

__device__ __forceinline__ u16 f2b(float f) {          // f32 -> bf16 (RNE)
  u32 u = __float_as_uint(f);
  u += 0x7FFFu + ((u >> 16) & 1u);
  return (u16)(u >> 16);
}
__device__ __forceinline__ float b2f(u16 h) {
  return __uint_as_float(((u32)h) << 16);
}

// ---------------------------------------------------------------------------
// Weight prep: f32 W[K,128] -> bf16 W^T[128,Kpad] (zero-padded K)
// ---------------------------------------------------------------------------
__global__ void prep_weight_kernel(const float* __restrict__ W,
                                   u16* __restrict__ WT, int K, int Kpad) {
  int i = blockIdx.x * blockDim.x + threadIdx.x;
  if (i >= LAT * Kpad) return;
  int n = i / Kpad, k = i - n * Kpad;
  float v = (k < K) ? W[(size_t)k * LAT + n] : 0.f;
  WT[i] = f2b(v);
}

// ---------------------------------------------------------------------------
// WMMA GEMM: out[M,128] = act( A @ W + bias ), A bf16 (gathered), W^T in LDS.
// Block = 256 thr = 8 waves; block tile 256x128; wave tile 32x128:
//   2 A fragments, 8 shared B fragments (register-resident), 16 C fragments.
// A fragment (16-bit A 16x32 layout): lane row = lane&15,
//   half 0: K {k..k+7, k+16..k+23}, half 1: K {k+8..k+15, k+24..k+31}.
// B fragment: lane col = lane&15, half selects K as in the A fragment above.
// C fragment: reg r -> row r + 8*half, col = lane&15.
// Weight staging uses GLOBAL_LOAD_ASYNC_TO_LDS_B128 (ASYNCcnt) per cdna5
// CDNA5 ISA; chunk count is an exact multiple of 256 threads so all
// lanes issue with full EXEC.
// ---------------------------------------------------------------------------
template<int KTOT, int MODE>
__global__ __launch_bounds__(256, 2) void wmma_gemm_kernel(
    const u16* __restrict__ A0, const u16* __restrict__ A1,
    const u16* __restrict__ A2,
    const int* __restrict__ snd, const int* __restrict__ rcv,
    const u16* __restrict__ WT, const float* __restrict__ bias,
    u16* __restrict__ out, int M, int relu)
{
  extern __shared__ u32 s_w[];           // W^T staged: 128*KTOT bf16
  constexpr int NKB  = KTOT / 32;        // 32-wide K blocks
  constexpr int KROW = KTOT / 2;         // u32 per W^T row
  const int tid  = threadIdx.x;
  const int lane = tid & 31;
  const int wv   = tid >> 5;
  const int half = lane >> 4;
  const int l16  = lane & 15;

  { // async stage of W^T into LDS: 16B chunks, ASYNCcnt-tracked
    constexpr int NCH = (LAT * KTOT * 2) / 16;     // multiple of 256
    u32 ldsoff = (u32)__builtin_amdgcn_groupstaticsize() + (u32)(tid * 16);
    u32 goff   = (u32)(tid * 16);
#pragma unroll 1
    for (int i = tid; i < NCH; i += 256) {
      asm volatile("global_load_async_to_lds_b128 %0, %1, %2"
                   :: "v"(ldsoff), "v"(goff), "s"(WT) : "memory");
      ldsoff += 256 * 16;
      goff   += 256 * 16;
    }
    asm volatile("s_wait_asynccnt 0x0" ::: "memory");
  }
  __syncthreads();

  const int m_base = blockIdx.x * 256 + wv * 32;
  int mA0 = m_base + l16;       if (mA0 >= M) mA0 = M - 1;  // clamp loads
  int mA1 = m_base + 16 + l16;  if (mA1 >= M) mA1 = M - 1;

  const u16 *r0p0, *r0p1, *r0p2, *r1p0, *r1p1, *r1p2;
  if constexpr (MODE == GM_PLAIN) {
    r0p0 = A0 + (size_t)mA0 * KTOT; r0p1 = r0p0; r0p2 = r0p0;
    r1p0 = A0 + (size_t)mA1 * KTOT; r1p1 = r1p0; r1p2 = r1p0;
  } else if constexpr (MODE == GM_EDGE) {
    r0p0 = A0 + (size_t)rcv[mA0] * LAT;  // x[receivers]
    r0p1 = A1 + (size_t)snd[mA0] * LAT;  // x[senders]
    r0p2 = A2 + (size_t)mA0 * LAT;       // e
    r1p0 = A0 + (size_t)rcv[mA1] * LAT;
    r1p1 = A1 + (size_t)snd[mA1] * LAT;
    r1p2 = A2 + (size_t)mA1 * LAT;
  } else { // GM_NODE
    r0p0 = A0 + (size_t)mA0 * LAT;       // x
    r0p1 = A1 + (size_t)mA0 * LAT;       // agg
    r0p2 = r0p0;
    r1p0 = A0 + (size_t)mA1 * LAT;
    r1p1 = A1 + (size_t)mA1 * LAT;
    r1p2 = r1p0;
  }

  union AB { v16bf v; v4u q[2]; };
  v8f c0[8] = {};
  v8f c1[8] = {};

#pragma unroll
  for (int kb = 0; kb < NKB; ++kb) {
    const int k0 = kb * 32;
    const u16 *rp0, *rp1;
    int kl;
    if constexpr (MODE == GM_PLAIN) { rp0 = r0p0; rp1 = r1p0; kl = k0; }
    else {
      const int s = k0 >> 7;             // constant under unroll
      rp0 = (s == 0) ? r0p0 : ((s == 1) ? r0p1 : r0p2);
      rp1 = (s == 0) ? r1p0 : ((s == 1) ? r1p1 : r1p2);
      kl = k0 & 127;
    }
    AB a0, a1;
    a0.q[0] = *(const v4u*)(rp0 + kl + (half << 3));       // K base .. +7
    a0.q[1] = *(const v4u*)(rp0 + kl + 16 + (half << 3));  // K base+16 .. +23
    a1.q[0] = *(const v4u*)(rp1 + kl + (half << 3));
    a1.q[1] = *(const v4u*)(rp1 + kl + 16 + (half << 3));

    const int kldsw = (k0 >> 1) + (half << 2);             // u32 offset in row
    AB b[8];
#pragma unroll
    for (int n = 0; n < 8; ++n) {                          // register-resident
      const u32* lp = &s_w[(n * 16 + l16) * KROW + kldsw]; // shared B frags
      b[n].q[0] = *(const v4u*)lp;
      b[n].q[1] = *(const v4u*)(lp + 8);
    }
#pragma unroll
    for (int n = 0; n < 8; ++n) {                          // back-to-back MACs
      c0[n] = __builtin_amdgcn_wmma_f32_16x16x32_bf16(
          false, a0.v, false, b[n].v, (short)0, c0[n], false, false);
      c1[n] = __builtin_amdgcn_wmma_f32_16x16x32_bf16(
          false, a1.v, false, b[n].v, (short)0, c1[n], false, false);
    }
  }

#pragma unroll
  for (int n = 0; n < 8; ++n) {
    const float bn = bias[n * 16 + l16];
#pragma unroll
    for (int r = 0; r < 8; ++r) {
      const int mrow0 = m_base + r + (half << 3);
      if (mrow0 < M) {
        float y = c0[n][r] + bn;
        if (relu) y = fmaxf(y, 0.f);
        out[(size_t)mrow0 * LAT + n * 16 + l16] = f2b(y);
      }
      const int mrow1 = m_base + 16 + r + (half << 3);
      if (mrow1 < M) {
        float y = c1[n][r] + bn;
        if (relu) y = fmaxf(y, 0.f);
        out[(size_t)mrow1 * LAT + n * 16 + l16] = f2b(y);
      }
    }
  }
}

// ---------------------------------------------------------------------------
// LayerNorm (+residual variants). One wave per 128-wide row.
// ---------------------------------------------------------------------------
__global__ __launch_bounds__(256) void layernorm_kernel(
    const u16* __restrict__ in, const float* __restrict__ g,
    const float* __restrict__ bb, float* __restrict__ stateF,
    u16* __restrict__ stateB, u16* __restrict__ enewB, int M, int op)
{
  const int row = blockIdx.x * 8 + (threadIdx.x >> 5);
  const int lane = threadIdx.x & 31;
  if (row >= M) return;
  const u16* p = in + (size_t)row * LAT + lane * 4;
  float v[4];
#pragma unroll
  for (int j = 0; j < 4; ++j) v[j] = b2f(p[j]);
  float s  = v[0] + v[1] + v[2] + v[3];
  float ss = v[0]*v[0] + v[1]*v[1] + v[2]*v[2] + v[3]*v[3];
  for (int off = 16; off > 0; off >>= 1) {
    s  += __shfl_xor(s,  off, 32);
    ss += __shfl_xor(ss, off, 32);
  }
  const float mean = s * (1.f / 128.f);
  const float inv  = rsqrtf(ss * (1.f / 128.f) - mean * mean + 1e-5f);
#pragma unroll
  for (int j = 0; j < 4; ++j) {
    const int c = lane * 4 + j;
    const size_t o = (size_t)row * LAT + c;
    const float y = (v[j] - mean) * inv * g[c] + bb[c];
    if (op == LN_SET) {
      stateF[o] = y; stateB[o] = f2b(y);
    } else if (op == LN_EDGE) {
      enewB[o] = f2b(y);
      const float t = stateF[o] + y; stateF[o] = t; stateB[o] = f2b(t);
    } else {
      const float t = stateF[o] + y; stateF[o] = t; stateB[o] = f2b(t);
    }
  }
}

// segment_sum: receivers = repeat(arange(N),16) -> contiguous 16-row blocks
__global__ void seg_sum_kernel(const u16* __restrict__ enew,
                               u16* __restrict__ agg, int N) {
  int i = blockIdx.x * blockDim.x + threadIdx.x;
  if (i >= N * LAT) return;
  const int n = i >> 7, c = i & 127;
  const u16* p = enew + (size_t)n * 16 * LAT + c;
  float s = 0.f;
#pragma unroll
  for (int j = 0; j < 16; ++j) s += b2f(p[(size_t)j * LAT]);
  agg[i] = f2b(s);
}

// node features: [norm vel (10), clipped bounds (4), type emb (16), pad(2)]
__global__ void node_feat_kernel(const float* __restrict__ pos,
    const float* __restrict__ noise, const int* __restrict__ ptype,
    const float* __restrict__ temb, const float* __restrict__ vmean,
    const float* __restrict__ vstd, u16* __restrict__ nf,
    float* __restrict__ mr, int N)
{
  int i = blockIdx.x * blockDim.x + threadIdx.x;
  if (i >= N) return;
  float p[6][2];
#pragma unroll
  for (int t = 0; t < 6; ++t)
#pragma unroll
    for (int d = 0; d < 2; ++d)
      p[t][d] = pos[(size_t)i*12 + t*2 + d] + noise[(size_t)i*12 + t*2 + d];
  mr[i*2+0] = p[5][0]; mr[i*2+1] = p[5][1];
  u16 o[32]; int j = 0;
#pragma unroll
  for (int t = 1; t < 6; ++t)
#pragma unroll
    for (int d = 0; d < 2; ++d)
      o[j++] = f2b((p[t][d] - p[t-1][d] - vmean[d]) / vstd[d]);
#pragma unroll
  for (int d = 0; d < 2; ++d)
    o[j++] = f2b(fminf(fmaxf(p[5][d] * (1.f/RCONN), -1.f), 1.f));
#pragma unroll
  for (int d = 0; d < 2; ++d)
    o[j++] = f2b(fminf(fmaxf((1.f - p[5][d]) * (1.f/RCONN), -1.f), 1.f));
  const int ty = ptype[i];
#pragma unroll
  for (int k = 0; k < 16; ++k) o[j++] = f2b(temb[ty*16 + k]);
  o[30] = 0; o[31] = 0;
  u16* dst = nf + (size_t)i * 32;
#pragma unroll
  for (int k = 0; k < 32; ++k) dst[k] = o[k];
}

// edge features: [disp/R (2), dist (1), pad to 32]
__global__ void edge_feat_kernel(const float* __restrict__ mr,
    const int* __restrict__ snd, const int* __restrict__ rcv,
    u16* __restrict__ ef, int E)
{
  int i = blockIdx.x * blockDim.x + threadIdx.x;
  if (i >= E) return;
  const int s = snd[i], r = rcv[i];
  const float dx = (mr[s*2+0] - mr[r*2+0]) * (1.f/RCONN);
  const float dy = (mr[s*2+1] - mr[r*2+1]) * (1.f/RCONN);
  const float dist = sqrtf(dx*dx + dy*dy);
  u16* dst = ef + (size_t)i * 32;
  dst[0] = f2b(dx); dst[1] = f2b(dy); dst[2] = f2b(dist);
#pragma unroll
  for (int k = 3; k < 32; ++k) dst[k] = 0;
}

// decoder last layer: [N,128] bf16 @ W[128,2] + b -> pred_acc f32
__global__ void dec_final_kernel(const u16* __restrict__ h,
    const float* __restrict__ W, const float* __restrict__ b,
    float* __restrict__ out, int N)
{
  int i = blockIdx.x * blockDim.x + threadIdx.x;
  if (i >= N * 2) return;
  const int n = i >> 1, d = i & 1;
  const u16* hp = h + (size_t)n * LAT;
  float acc = b[d];
  for (int k = 0; k < LAT; ++k) acc += b2f(hp[k]) * W[k*2 + d];
  out[i] = acc;
}

// target = ((next+noise5) - 2*noisy5 + noisy4 - acc_mean)/acc_std
__global__ void target_kernel(const float* __restrict__ nextp,
    const float* __restrict__ noise, const float* __restrict__ pos,
    const float* __restrict__ accm, const float* __restrict__ accs,
    float* __restrict__ out, int N)
{
  int i = blockIdx.x * blockDim.x + threadIdx.x;
  if (i >= N * 2) return;
  const int n = i >> 1, d = i & 1;
  const float n5 = noise[(size_t)n*12 + 10 + d];
  const float p5 = pos[(size_t)n*12 + 10 + d] + n5;
  const float p4 = pos[(size_t)n*12 + 8 + d] + noise[(size_t)n*12 + 8 + d];
  const float acc = (nextp[i] + n5) - 2.f * p5 + p4;
  out[i] = (acc - accm[d]) / accs[d];
}

// ---------------------------------------------------------------------------
// Host orchestration (all async on `stream`, graph-capture safe)
// ---------------------------------------------------------------------------
extern "C" void kernel_launch(void* const* d_in, const int* in_sizes, int n_in,
                              void* d_out, int out_size, void* d_ws, size_t ws_size,
                              hipStream_t stream) {
  (void)in_sizes; (void)out_size; (void)ws_size;
  if (n_in < 49) return;
  const int N = 20000, E = 320000;

  auto F = [&](int i) { return (const float*)d_in[i]; };
  auto I = [&](int i) { return (const int*)d_in[i]; };
  const int* snd = I(4);
  const int* rcv = I(5);

  char* base = (char*)d_ws; size_t cur = 0;
  auto alloc = [&](size_t bytes) -> char* {
    char* p = base + cur; cur = (cur + bytes + 255) & ~(size_t)255; return p;
  };
  u16*   nf   = (u16*)  alloc((size_t)N * 32 * 2);
  u16*   ef   = (u16*)  alloc((size_t)E * 32 * 2);
  float* mr   = (float*)alloc((size_t)N * 2 * 4);
  float* xF   = (float*)alloc((size_t)N * LAT * 4);
  u16*   xB   = (u16*)  alloc((size_t)N * LAT * 2);
  float* eF   = (float*)alloc((size_t)E * LAT * 4);
  u16*   eB   = (u16*)  alloc((size_t)E * LAT * 2);
  u16*   aggB = (u16*)  alloc((size_t)N * LAT * 2);
  u16*   t1   = (u16*)  alloc((size_t)E * LAT * 2);
  u16*   t2   = (u16*)  alloc((size_t)E * LAT * 2);

  auto prep = [&](const float* W, int K, int Kpad) -> u16* {
    u16* wt = (u16*)alloc((size_t)LAT * Kpad * 2);
    const int tot = LAT * Kpad;
    prep_weight_kernel<<<(tot + 255) / 256, 256, 0, stream>>>(W, wt, K, Kpad);
    return wt;
  };

  u16* wt_en[3] = { prep(F(7), 30, 32),  prep(F(9), 128, 128),  prep(F(11), 128, 128) };
  u16* wt_ee[3] = { prep(F(15), 3, 32),  prep(F(17), 128, 128), prep(F(19), 128, 128) };
  u16* wt_pe[10][3]; u16* wt_pn[10][3];
  for (int s = 0; s < 10; ++s) {
    wt_pe[s][0] = prep(F(23) + (size_t)s * 384 * LAT, 384, 384);
    wt_pe[s][1] = prep(F(25) + (size_t)s * 128 * LAT, 128, 128);
    wt_pe[s][2] = prep(F(27) + (size_t)s * 128 * LAT, 128, 128);
    wt_pn[s][0] = prep(F(31) + (size_t)s * 256 * LAT, 256, 256);
    wt_pn[s][1] = prep(F(33) + (size_t)s * 128 * LAT, 128, 128);
    wt_pn[s][2] = prep(F(35) + (size_t)s * 128 * LAT, 128, 128);
  }
  u16* wt_d0 = prep(F(39), 128, 128);
  u16* wt_d1 = prep(F(41), 128, 128);

  node_feat_kernel<<<(N + 255) / 256, 256, 0, stream>>>(
      F(2), F(1), I(3), F(6), F(45), F(46), nf, mr, N);
  edge_feat_kernel<<<(E + 255) / 256, 256, 0, stream>>>(mr, snd, rcv, ef, E);

  auto gemm = [&](const u16* A0, const u16* A1, const u16* A2, const u16* WT,
                  const float* bias, u16* out, int M, int Ktot, int mode, int relu) {
    dim3 g((M + 255) / 256);
    size_t sh = (size_t)LAT * Ktot * 2;
    if (mode == GM_EDGE)
      wmma_gemm_kernel<384, GM_EDGE><<<g, 256, sh, stream>>>(
          A0, A1, A2, snd, rcv, WT, bias, out, M, relu);
    else if (mode == GM_NODE)
      wmma_gemm_kernel<256, GM_NODE><<<g, 256, sh, stream>>>(
          A0, A1, A2, snd, rcv, WT, bias, out, M, relu);
    else if (Ktot == 32)
      wmma_gemm_kernel<32, GM_PLAIN><<<g, 256, sh, stream>>>(
          A0, A1, A2, snd, rcv, WT, bias, out, M, relu);
    else
      wmma_gemm_kernel<128, GM_PLAIN><<<g, 256, sh, stream>>>(
          A0, A1, A2, snd, rcv, WT, bias, out, M, relu);
  };
  auto ln = [&](const u16* in, const float* g, const float* b, float* sF,
                u16* sB, u16* enew, int M, int op) {
    layernorm_kernel<<<dim3((M + 7) / 8), dim3(256), 0, stream>>>(
        in, g, b, sF, sB, enew, M, op);
  };

  // ---- encoders ----
  gemm(ef, ef, ef, wt_ee[0], F(16), t1, E, 32,  GM_PLAIN, 1);
  gemm(t1, t1, t1, wt_ee[1], F(18), t2, E, 128, GM_PLAIN, 1);
  gemm(t2, t2, t2, wt_ee[2], F(20), t1, E, 128, GM_PLAIN, 0);
  ln(t1, F(21), F(22), eF, eB, nullptr, E, LN_SET);

  gemm(nf, nf, nf, wt_en[0], F(8),  t1, N, 32,  GM_PLAIN, 1);
  gemm(t1, t1, t1, wt_en[1], F(10), t2, N, 128, GM_PLAIN, 1);
  gemm(t2, t2, t2, wt_en[2], F(12), t1, N, 128, GM_PLAIN, 0);
  ln(t1, F(13), F(14), xF, xB, nullptr, N, LN_SET);

  // ---- 10 processor steps ----
  for (int s = 0; s < 10; ++s) {
    gemm(xB, xB, eB, wt_pe[s][0], F(24) + s * LAT, t1, E, 384, GM_EDGE, 1);
    gemm(t1, t1, t1, wt_pe[s][1], F(26) + s * LAT, t2, E, 128, GM_PLAIN, 1);
    gemm(t2, t2, t2, wt_pe[s][2], F(28) + s * LAT, t1, E, 128, GM_PLAIN, 0);
    ln(t1, F(29) + s * LAT, F(30) + s * LAT, eF, eB, t2, E, LN_EDGE);
    seg_sum_kernel<<<((N * LAT) + 255) / 256, 256, 0, stream>>>(t2, aggB, N);
    gemm(xB, aggB, xB, wt_pn[s][0], F(32) + s * LAT, t1, N, 256, GM_NODE, 1);
    gemm(t1, t1, t1, wt_pn[s][1], F(34) + s * LAT, t2, N, 128, GM_PLAIN, 1);
    gemm(t2, t2, t2, wt_pn[s][2], F(36) + s * LAT, t1, N, 128, GM_PLAIN, 0);
    ln(t1, F(37) + s * LAT, F(38) + s * LAT, xF, xB, nullptr, N, LN_NODE);
  }

  // ---- decoder + target ----
  gemm(xB, xB, xB, wt_d0, F(40), t1, N, 128, GM_PLAIN, 1);
  gemm(t1, t1, t1, wt_d1, F(42), t2, N, 128, GM_PLAIN, 1);
  dec_final_kernel<<<((N * 2) + 255) / 256, 256, 0, stream>>>(
      t2, F(43), F(44), (float*)d_out, N);
  target_kernel<<<((N * 2) + 255) / 256, 256, 0, stream>>>(
      F(0), F(1), F(2), F(47), F(48), (float*)d_out + (size_t)N * 2, N);
}
